// IAGNN_18734647345349
// MI455X (gfx1250) — compile-verified, matrix operand order checked by reading, not run on hardware
//
#include <hip/hip_runtime.h>

#define E_N 8000
#define R_N 32
#define D_N 128
#define LEAKY 0.2f
#define SPITCH 144   // bf16 elements per staged row (128 + 16 pad, keeps 32B alignment)

// Verified to assemble on this toolchain (round 2 compiled OK with it enabled).
#define USE_ASYNC_LDS 1

typedef __attribute__((ext_vector_type(16))) __bf16 v16bf;
typedef __attribute__((ext_vector_type(8)))  float  v8f;

#ifndef __has_builtin
#define __has_builtin(x) 0
#endif
#if __has_builtin(__builtin_amdgcn_wmma_f32_16x16x32_bf16)
#define HAS_WMMA_BF16 1
#else
#define HAS_WMMA_BF16 0
#endif

// Async copy 16B from global to LDS (per-lane), tracked by ASYNCcnt.
__device__ __forceinline__ void async_g2l_b128(const float* g, float* l) {
#if USE_ASYNC_LDS
    unsigned lds_off = (unsigned)(unsigned long long)(uintptr_t)l;   // low 32 bits = LDS byte addr
    unsigned long long ga = (unsigned long long)(uintptr_t)g;
    asm volatile("global_load_async_to_lds_b128 %0, %1, off" :: "v"(lds_off), "v"(ga) : "memory");
#else
    const float4* gs = (const float4*)g;
    *(float4*)l = *gs;
#endif
}
__device__ __forceinline__ void async_wait0() {
#if USE_ASYNC_LDS
    asm volatile("s_wait_asynccnt 0x0" ::: "memory");
#endif
}

// ---------------- K0: softmax(adj_agg, axis=1), one wave per entity ----------------
__global__ void k0_adj_softmax(const float* __restrict__ adj, float* __restrict__ adj_s) {
    int wave = threadIdx.x >> 5, lane = threadIdx.x & 31;
    int e = blockIdx.x * 8 + wave;
    float a = adj[e * R_N + lane];
    float m = a;
#pragma unroll
    for (int off = 16; off > 0; off >>= 1) m = fmaxf(m, __shfl_xor(m, off, 32));
    float ex = __expf(a - m);
    float s = ex;
#pragma unroll
    for (int off = 16; off > 0; off >>= 1) s += __shfl_xor(s, off, 32);
    adj_s[e * R_N + lane] = ex / s;
}

// ---------------- K1: per-block column sums over entities (no atomics) ----------------
// 64 blocks x 128 threads (thread = d), 125 entities each. Accumulates
// sum_e exp(att[e,r,d]) and sum_e att_R[e,r,d] in registers.
__global__ void __attribute__((amdgpu_waves_per_eu(1)))
k1_colsums(const float* __restrict__ h, const float* __restrict__ rg,
           const float* __restrict__ adj_s,
           float* __restrict__ partE, float* __restrict__ partR) {
    __shared__ float rmat[R_N * D_N];
    __shared__ float adjrow[R_N];
    int tid = threadIdx.x;
#pragma unroll
    for (int k = 0; k < 8; k++) {                 // 128 threads x 8 x 16B = 16KB
        int i = k * 512 + tid * 4;
        async_g2l_b128(rg + i, rmat + i);
    }
    async_wait0();
    __syncthreads();
    float colE[32], colR[32];
#pragma unroll
    for (int r2 = 0; r2 < 32; r2++) { colE[r2] = 0.f; colR[r2] = 0.f; }
    int e0 = blockIdx.x * 125;
#pragma unroll 1
    for (int k = 0; k < 125; k++) {
        int e = e0 + k;
        if (tid < 32) adjrow[tid] = adj_s[e * R_N + tid];
        __syncthreads();
        float hv = h[e * D_N + tid];
        if (k + 1 < 125) __builtin_prefetch(&h[(e + 1) * D_N + tid], 0, 0);
        float t[32]; float m = -3.0e38f;
#pragma unroll
        for (int r2 = 0; r2 < 32; r2++) {
            float v = adjrow[r2] * hv * rmat[r2 * D_N + tid];
            v = v >= 0.f ? v : LEAKY * v;
            t[r2] = v; m = fmaxf(m, v);
        }
        float s = 0.f;
#pragma unroll
        for (int r2 = 0; r2 < 32; r2++) { float er = __expf(t[r2] - m); t[r2] = er; s += er; }
        float em = __expf(m);          // exp(att) = er * exp(m); att magnitudes <~20, safe in f32
        float invs = 1.0f / s;
#pragma unroll
        for (int r2 = 0; r2 < 32; r2++) { colE[r2] += t[r2] * em; colR[r2] += t[r2] * invs; }
        __syncthreads();
    }
    int b = blockIdx.x;
#pragma unroll
    for (int r2 = 0; r2 < 32; r2++) {
        partE[b * 4096 + r2 * D_N + tid] = colE[r2];
        partR[b * 4096 + r2 * D_N + tid] = colR[r2];
    }
}

// ---------------- K2: reduce 64 partials; emit colE totals and r_prime ----------------
__global__ void k2_reduce(const float* __restrict__ partE, const float* __restrict__ partR,
                          const float* __restrict__ rg, float* __restrict__ colE,
                          float* __restrict__ out_rprime) {
    int j = blockIdx.x * 128 + threadIdx.x;   // 0..4095 (= r*128+d)
    float sE = 0.f, sR = 0.f;
    for (int b = 0; b < 64; b++) { sE += partE[b * 4096 + j]; sR += partR[b * 4096 + j]; }
    colE[j] = sE;
    out_rprime[j] = rg[j] * sR;               // r_prime[r,d] = r[r,d] * sum_e att_R
}

// ---------------- K3: h_prime + alpha GEMV via WMMA (wave-per-entity) ----------------
__global__ void __attribute__((amdgpu_waves_per_eu(1)))
k3_main(const float* __restrict__ h, const float* __restrict__ rg,
        const float* __restrict__ adj_s, const float* __restrict__ colE,
        const float* __restrict__ wl, const float* __restrict__ bl,
        float* __restrict__ out_h, float* __restrict__ out_alpha) {
    __shared__ float rmat[R_N * D_N];
    __shared__ float cinv[R_N * D_N];
    __shared__ float wlds[D_N];
    __shared__ float adjW[4][R_N];
    __shared__ __bf16 Sb[4][R_N * SPITCH];   // per-wave staging of att_R * w in bf16
    int tid = threadIdx.x, w = tid >> 5, lane = tid & 31;
#pragma unroll
    for (int k = 0; k < 8; k++) {                 // async-stage r matrix (16KB)
        int i = k * 512 + tid * 4;
        async_g2l_b128(rg + i, rmat + i);
    }
    for (int i = tid; i < R_N * D_N; i += 128) cinv[i] = 1.0f / colE[i];
    if (tid < D_N) wlds[tid] = wl[tid];
    async_wait0();
    __syncthreads();
    float bias = bl[0];
    v16bf ones;
#pragma unroll
    for (int j = 0; j < 16; j++) ones[j] = (__bf16)1.0f;

#pragma unroll 1
    for (int it = 0; it < 4; it++) {
        int e = blockIdx.x * 16 + w * 4 + it;
        adjW[w][lane] = adj_s[e * R_N + lane];
        asm volatile("s_wait_dscnt 0x0" ::: "memory");   // wave-local LDS visibility
#pragma unroll 1
        for (int i = 0; i < 4; i++) {
            int d = i * 32 + lane;
            float hv = h[e * D_N + d];
            float t[32]; float m = -3.0e38f;
#pragma unroll
            for (int r2 = 0; r2 < 32; r2++) {
                float v = adjW[w][r2] * hv * rmat[r2 * D_N + d];
                v = v >= 0.f ? v : LEAKY * v;
                t[r2] = v; m = fmaxf(m, v);
            }
            float s = 0.f;
#pragma unroll
            for (int r2 = 0; r2 < 32; r2++) { float er = __expf(t[r2] - m); t[r2] = er; s += er; }
            float em = __expf(m);
            float hsum = 0.f;
#pragma unroll
            for (int r2 = 0; r2 < 32; r2++) hsum += t[r2] * em * cinv[r2 * D_N + d];
            out_h[e * D_N + d] = hv * hsum;              // h_prime
            float invs = 1.0f / s; float wd = wlds[d];
#pragma unroll
            for (int r2 = 0; r2 < 32; r2++)
                Sb[w][r2 * SPITCH + i * 32 + lane] = (__bf16)(t[r2] * invs * wd);
        }
        asm volatile("s_wait_dscnt 0x0" ::: "memory");
#if HAS_WMMA_BF16
#pragma unroll
        for (int rblk = 0; rblk < 2; rblk++) {
            v8f acc = {0.f, 0.f, 0.f, 0.f, 0.f, 0.f, 0.f, 0.f};
            int row = rblk * 16 + (lane & 15);       // A row = lane%16 (documented M mapping)
            int dhalf = (lane >> 4) << 4;            // lane pair covers the 32-wide K chunk
#pragma unroll
            for (int kc = 0; kc < 4; kc++) {
                const __bf16* sp = &Sb[w][row * SPITCH + kc * 32 + dhalf];
                v16bf a;
#pragma unroll
                for (int j = 0; j < 16; j++) a[j] = sp[j];
                // B == all-ones -> D[m,n] = sum over the row's 32 slots (K-slot-order safe)
                acc = __builtin_amdgcn_wmma_f32_16x16x32_bf16(
                    false, a, false, ones, (short)0, acc, false, false);
            }
            if (lane == 0 || lane == 16) {           // D layout: vgpr i = M=i (lane0) / M=8+i (lane16)
                int mb = rblk * 16 + (lane >> 4) * 8;
#pragma unroll
                for (int j2 = 0; j2 < 8; j2++) out_alpha[e * R_N + mb + j2] = acc[j2] + bias;
            }
        }
#else
        if (lane < 32) {   // fallback: serial row reduction
            float a = 0.f;
            for (int d2 = 0; d2 < D_N; d2++) a += (float)Sb[w][lane * SPITCH + d2];
            out_alpha[e * R_N + lane] = a + bias;
        }
#endif
    }
}

// ---------------- K4: softmax over each 8000-chunk of the flat alpha (raw reshape) ----------------
__global__ void k4_alpha_softmax(float* __restrict__ alpha) {
    __shared__ float red[256];
    float* base = alpha + blockIdx.x * E_N;
    int tid = threadIdx.x;
    float m = -3.0e38f;
    for (int i = tid; i < E_N; i += 256) m = fmaxf(m, base[i]);
    red[tid] = m; __syncthreads();
    for (int o = 128; o > 0; o >>= 1) { if (tid < o) red[tid] = fmaxf(red[tid], red[tid + o]); __syncthreads(); }
    m = red[0]; __syncthreads();
    float s = 0.f;
    for (int i = tid; i < E_N; i += 256) s += __expf(base[i] - m);
    red[tid] = s; __syncthreads();
    for (int o = 128; o > 0; o >>= 1) { if (tid < o) red[tid] += red[tid + o]; __syncthreads(); }
    s = red[0]; __syncthreads();
    float inv = 1.0f / s;
    for (int i = tid; i < E_N; i += 256) base[i] = __expf(base[i] - m) * inv;
}

extern "C" void kernel_launch(void* const* d_in, const int* in_sizes, int n_in,
                              void* d_out, int out_size, void* d_ws, size_t ws_size,
                              hipStream_t stream) {
    (void)in_sizes; (void)n_in; (void)out_size; (void)ws_size;
    const float* h   = (const float*)d_in[0];   // [8000,128]
    const float* rg  = (const float*)d_in[1];   // [32,128]
    const float* adj = (const float*)d_in[2];   // [8000,32]
    const float* wl  = (const float*)d_in[3];   // [128]
    const float* bl  = (const float*)d_in[4];   // [1]
    float* out   = (float*)d_out;
    float* out_h = out;                          // [8000,128]
    float* out_r = out + E_N * D_N;              // [32,128]
    float* out_a = out + E_N * D_N + R_N * D_N;  // [32,8000,1] flat

    float* wsf   = (float*)d_ws;                 // ~3.1 MB total
    float* adj_s = wsf;                          // 256000
    float* colE  = wsf + 256000;                 // 4096
    float* partE = wsf + 260096;                 // 64*4096
    float* partR = wsf + 522240;                 // 64*4096

    k0_adj_softmax<<<E_N / 8, 256, 0, stream>>>(adj, adj_s);
    k1_colsums   <<<64, 128, 0, stream>>>(h, rg, adj_s, partE, partR);
    k2_reduce    <<<32, 128, 0, stream>>>(partE, partR, rg, colE, out_r);
    k3_main      <<<E_N / 16, 128, 0, stream>>>(h, rg, adj_s, colE, wl, bl, out_h, out_a);
    k4_alpha_softmax<<<R_N, 256, 0, stream>>>(out_a);
}
